// PDecoder_19894288515346
// MI455X (gfx1250) — compile-verified, matrix operand order checked by reading, not run on hardware
//
#include <hip/hip_runtime.h>
#include <math.h>
#include <stdint.h>

#define BATCH 128
#define LSEQ  256
#define EMB   512
#define HID   512

#define USE_ASYNC_LDS 1

typedef __attribute__((ext_vector_type(16))) __bf16 v16bf;
typedef __attribute__((ext_vector_type(8)))  float  v8f;
typedef unsigned short ushort_t;

// ---------------- workspace layout (bytes) ----------------
#define CTXBF_OFF   0ull                 // 128*256*512 bf16 = 33554432
#define WIH_OFF     33554432ull          // 2048*512*2
#define WHH_OFF     35651584ull
#define WAIN_OFF    37748736ull          // 512*512*2
#define WHO_OFF     38273024ull          // 512*1024*2
#define WACTX_OFF   39321600ull
#define HBF_OFF     39845888ull          // 128*512*2
#define HLSTM_OFF   39976960ull
#define HS_OFF      40108032ull
#define DEC_OFF     40239104ull
#define INPBF_OFF   40370176ull
#define C_OFF       40501248ull          // 128*512*4
#define MASK_OFF    40763392ull          // 128*256*4
#define ATT_OFF     40894464ull
#define BAR_OFF     41025536ull          // 128 * u32
#define CTXIN_OFF   41026048ull          // optional: context as bf16 (33554432)
#define WS_NEED_BIG (CTXIN_OFF + 33554432ull)

// ---------------- d_out layout (float elements) ----------------
#define OUT_ALPHA   0
#define OUT_PTR     8388608              // B*L*L
#define OUT_HF      8421376              // + B*L
#define OUT_CF      8486912              // + B*H

#define NEG_BIG (-3.0e38f)

__device__ __forceinline__ ushort_t f2bf(float f){
  union { __bf16 b; ushort_t u; } cv;
  cv.b = (__bf16)f;                      // native v_cvt to bf16 (RNE)
  return cv.u;
}
__device__ __forceinline__ float bf2f(ushort_t h){
  return __uint_as_float(((unsigned)h) << 16);
}
__device__ __forceinline__ float fsigmoid(float x){
  return 1.0f / (1.0f + __expf(-x));
}
__device__ __forceinline__ float ftanh(float x){
#if __has_builtin(__builtin_amdgcn_tanhf)
  return __builtin_amdgcn_tanhf(x);     // single v_tanh_f32 TRANS op on gfx1250
#else
  float e = __expf(2.0f * x);
  return 1.0f - 2.0f / (e + 1.0f);
#endif
}

__device__ __forceinline__ v8f wmma_bf16(v16bf a, v16bf b, v8f c){
  return __builtin_amdgcn_wmma_f32_16x16x32_bf16(false, a, false, b, (short)0, c, false, false);
}

// ---- async global -> LDS copy (ASYNCcnt-tracked, CDNA5) ----
#if USE_ASYNC_LDS
__device__ __forceinline__ void async_g2l_b128(void* lds, const void* gptr){
  unsigned ldsoff = (unsigned)(uintptr_t)lds;   // low 32 bits of generic = LDS offset
  asm volatile("global_load_async_to_lds_b128 %0, %1, off"
               :: "v"(ldsoff), "v"(gptr) : "memory");
}
__device__ __forceinline__ void wait_async(){
  asm volatile("s_wait_asynccnt 0" ::: "memory");
}
#else
__device__ __forceinline__ void async_g2l_b128(void* lds, const void* gptr){
  *(uint4*)lds = *(const uint4*)gptr;
}
__device__ __forceinline__ void wait_async(){}
#endif

// 16-bit A/B fragment from a row-major [rows, ldk] bf16 matrix.
// ISA layout: lane m (0..15 / 16..31 same m), u[0..7]=K[k0+half*8 ..],
// u[8..15]=K[k0+16+half*8 ..]  -> two 16B chunks per lane.
__device__ __forceinline__ v16bf load_frag(const ushort_t* M, int ldk, int row, int k0){
  int lane = threadIdx.x & 31;
  const ushort_t* p = M + (size_t)(row + (lane & 15)) * ldk + k0 + ((lane >> 4) << 3);
  union { v16bf v; uint4 q[2]; } u;
  u.q[0] = *(const uint4*)p;
  u.q[1] = *(const uint4*)(p + 16);
  return u.v;
}
// Same, but convert from f32 source on the fly (fallback when ws too small).
__device__ __forceinline__ v16bf load_frag_f32(const float* M, int ldk, int row, int k0){
  int lane = threadIdx.x & 31;
  const float* p = M + (size_t)(row + (lane & 15)) * ldk + k0 + ((lane >> 4) << 3);
  union { v16bf v; ushort_t us[16]; } u;
#pragma unroll
  for (int j = 0; j < 8; ++j) u.us[j]     = f2bf(p[j]);
#pragma unroll
  for (int j = 0; j < 8; ++j) u.us[8 + j] = f2bf(p[16 + j]);
  return u.v;
}

// group-wide (16 WG) arrive-and-spin barrier with monotonically increasing target
__device__ __forceinline__ void group_barrier(unsigned* bar, unsigned target){
  __threadfence();
  __syncthreads();
  if (threadIdx.x == 0){
    __hip_atomic_fetch_add(bar, 1u, __ATOMIC_RELEASE, __HIP_MEMORY_SCOPE_AGENT);
    while (__hip_atomic_load(bar, __ATOMIC_ACQUIRE, __HIP_MEMORY_SCOPE_AGENT) < target){
      __builtin_amdgcn_s_sleep(1);
    }
  }
  __syncthreads();
}

// ================= kernel 0: convert / init =================
__global__ void k_init(const float* Wih, const float* Whh, const float* Wain,
                       const float* Who, const float* Wactx,
                       const float* h0, const float* c0, const float* dec0,
                       const float* context,
                       ushort_t* wih_bf, ushort_t* whh_bf, ushort_t* wain_bf,
                       ushort_t* who_bf, ushort_t* wactx_bf,
                       ushort_t* h_bf, ushort_t* dec_bf, float* c_f,
                       float* mask, unsigned* bar, ushort_t* ctxin_bf)
{
  int stride = gridDim.x * blockDim.x;
  int tid = blockIdx.x * blockDim.x + threadIdx.x;
  for (int i = tid; i < 2048*512; i += stride){ wih_bf[i] = f2bf(Wih[i]); whh_bf[i] = f2bf(Whh[i]); }
  for (int i = tid; i < 512*1024; i += stride){ who_bf[i] = f2bf(Who[i]); }
  for (int i = tid; i < 512*512;  i += stride){ wain_bf[i] = f2bf(Wain[i]); wactx_bf[i] = f2bf(Wactx[i]); }
  for (int i = tid; i < 128*512;  i += stride){ h_bf[i] = f2bf(h0[i]); dec_bf[i] = f2bf(dec0[i]); c_f[i] = c0[i]; }
  for (int i = tid; i < 128*256;  i += stride){ mask[i] = 1.0f; }
  for (int i = tid; i < 128;      i += stride){ bar[i] = 0u; }
  if (ctxin_bf){
    for (int i = tid; i < 128*256*512; i += stride) ctxin_bf[i] = f2bf(context[i]);
  }
}

// ================= kernel 1: ctx = context @ Wa_ctx^T + ba_ctx =================
// M = B*L = 32768 rows, N = 512, K = 512.  One 16x16 tile per wave.
__global__ void __launch_bounds__(256) k_ctx_bf(const ushort_t* ctxin_bf, const ushort_t* wactx_bf,
                                                const float* ba_ctx, ushort_t* ctx_bf)
{
  int wave = threadIdx.x >> 5;
  int lane = threadIdx.x & 31;
  int tile = blockIdx.x * 8 + wave;          // 65536 tiles
  int mi = tile >> 5;                        // 0..2047
  int ni = tile & 31;                        // 0..31
  v8f acc = {0.f,0.f,0.f,0.f,0.f,0.f,0.f,0.f};
#pragma unroll 4
  for (int k = 0; k < 16; ++k){
    v16bf a = load_frag(ctxin_bf, 512, mi*16, k*32);
    v16bf b = load_frag(wactx_bf, 512, ni*16, k*32);
    acc = wmma_bf16(a, b, acc);
  }
  int n = ni*16 + (lane & 15);
  int half = lane >> 4;
  float bias = ba_ctx[n];
#pragma unroll
  for (int r = 0; r < 8; ++r){
    int m = mi*16 + r + 8*half;
    ctx_bf[(size_t)m*512 + n] = f2bf(acc[r] + bias);
  }
}

__global__ void __launch_bounds__(256) k_ctx_f32(const float* context, const ushort_t* wactx_bf,
                                                 const float* ba_ctx, ushort_t* ctx_bf)
{
  int wave = threadIdx.x >> 5;
  int lane = threadIdx.x & 31;
  int tile = blockIdx.x * 8 + wave;
  int mi = tile >> 5;
  int ni = tile & 31;
  v8f acc = {0.f,0.f,0.f,0.f,0.f,0.f,0.f,0.f};
#pragma unroll 4
  for (int k = 0; k < 16; ++k){
    v16bf a = load_frag_f32(context, 512, mi*16, k*32);
    v16bf b = load_frag(wactx_bf, 512, ni*16, k*32);
    acc = wmma_bf16(a, b, acc);
  }
  int n = ni*16 + (lane & 15);
  int half = lane >> 4;
  float bias = ba_ctx[n];
#pragma unroll
  for (int r = 0; r < 8; ++r){
    int m = mi*16 + r + 8*half;
    ctx_bf[(size_t)m*512 + n] = f2bf(acc[r] + bias);
  }
}

// ================= kernel 2: persistent decoder =================
// grid = 128 blocks = 8 batch-groups x 16 WGs; block = 256 threads (8 waves).
// group g handles batch rows [g*16, g*16+16); WG w owns h-columns [w*32, w*32+32)
// and attention positions [w*16, w*16+16).
__global__ void __launch_bounds__(256) k_decode(
    const float* embedded,
    const float* bih, const float* bhh, const float* bho,
    const float* ba_in, const float* Vvec,
    const ushort_t* ctx_bf,
    const ushort_t* wih_bf, const ushort_t* whh_bf,
    const ushort_t* wain_bf, const ushort_t* who_bf,
    ushort_t* h_bf, ushort_t* hlstm_bf, ushort_t* hs_bf, ushort_t* dec_bf,
    ushort_t* inp_bf,
    float* c_f, float* mask_g, float* att_g,
    unsigned* bar, float* out)
{
  __shared__ ushort_t sA[16*1024];      // 32KB: GEMM A staging; +8192 doubles as inp bf16
  __shared__ float    sRed[16*256];     // 16KB: gate buffer (phase A) / alpha (D,E,G)
  __shared__ float    sV[512];
  __shared__ float    sRmax[16], sRsum[16];
  __shared__ int      sIdx[16];

  const int tid  = threadIdx.x;
  const int lane = tid & 31;
  const int wv   = tid >> 5;               // wave 0..7
  const int grp  = blockIdx.x >> 4;        // batch group 0..7
  const int w    = blockIdx.x & 15;        // WG within group
  const int b0   = grp * 16;
  const int hw   = w * 32;
  unsigned* barp = bar + grp * 16;
  unsigned target = 0;

  for (int i = tid; i < 512; i += 256) sV[i] = Vvec[i];
  __syncthreads();

  for (int t = 0; t < LSEQ; ++t){
    // ---------- Phase A: gates = [dec|h] @ [Wih|Whh]^T + b ; LSTM cell ----------
#pragma unroll
    for (int it = 0; it < 8; ++it){
      int id = tid + it*256;               // 2048 x b128 = 16 rows x 128
      int row = id >> 7, q = id & 127;
      int b = b0 + row;
      const ushort_t* src = (q < 64) ? (dec_bf + b*512 + q*8)
                                     : (h_bf  + b*512 + (q-64)*8);
      async_g2l_b128(&sA[row*1024 + q*8], src);
    }
    wait_async();
    __syncthreads();
    {
      int g    = wv >> 1;                  // gate 0..3 (i,f,g,o)
      int coln = (wv & 1) * 16;            // sub-column within WG's 32-col slice
      int nrow = g*512 + hw + coln;
      const ushort_t* wrow_ih = wih_bf + (size_t)(nrow + (lane & 15)) * 512 + ((lane >> 4) << 3);
      const ushort_t* wrow_hh = whh_bf + (size_t)(nrow + (lane & 15)) * 512 + ((lane >> 4) << 3);
      v8f acc = {0.f,0.f,0.f,0.f,0.f,0.f,0.f,0.f};
      for (int k = 0; k < 32; ++k){
        int kk = k*32;
        if (kk + 128 < 512)       __builtin_prefetch(wrow_ih + kk + 128, 0, 3);
        else if (kk + 128 < 1024) __builtin_prefetch(wrow_hh + kk + 128 - 512, 0, 3);
        v16bf a = load_frag(sA, 1024, 0, kk);
        v16bf b = (kk < 512) ? load_frag(wih_bf, 512, nrow, kk)
                             : load_frag(whh_bf, 512, nrow, kk - 512);
        acc = wmma_bf16(a, b, acc);
      }
      int n = nrow + (lane & 15);
      float bias = bih[n] + bhh[n];
      int half = lane >> 4;
      int cloc = coln + (lane & 15);
#pragma unroll
      for (int r = 0; r < 8; ++r){
        int m = r + 8*half;
        sRed[g*512 + m*32 + cloc] = acc[r] + bias;
      }
    }
    __syncthreads();
#pragma unroll
    for (int it = 0; it < 2; ++it){
      int id = tid + it*256;               // 16m x 32h
      int m = id >> 5, hh = id & 31;
      int b = b0 + m, hc = hw + hh;
      float ig = fsigmoid(sRed[0*512 + m*32 + hh]);
      float fg = fsigmoid(sRed[1*512 + m*32 + hh]);
      float gg = ftanh   (sRed[2*512 + m*32 + hh]);
      float og = fsigmoid(sRed[3*512 + m*32 + hh]);
      float cn = fg * c_f[b*512 + hc] + ig * gg;
      c_f[b*512 + hc] = cn;
      if (t == LSEQ-1) out[OUT_CF + b*512 + hc] = cn;
      float hl = og * ftanh(cn);
      hlstm_bf[b*512 + hc] = f2bf(hl);
    }
    target += 16; group_barrier(barp, target);               // B1

    // ---------- Phase B: inp = h_lstm @ Wa_in^T + ba_in ----------
#pragma unroll
    for (int it = 0; it < 4; ++it){
      int id = tid + it*256;               // 1024 x b128 = 16 rows x 64
      int row = id >> 6, q = id & 63;
      async_g2l_b128(&sA[row*512 + q*8], hlstm_bf + (b0+row)*512 + q*8);
    }
    wait_async();
    __syncthreads();
    if (wv < 2){
      int coln = hw + wv*16;
      v8f acc = {0.f,0.f,0.f,0.f,0.f,0.f,0.f,0.f};
      for (int k = 0; k < 16; ++k){
        v16bf a = load_frag(sA, 512, 0, k*32);
        v16bf b = load_frag(wain_bf, 512, coln, k*32);
        acc = wmma_bf16(a, b, acc);
      }
      int n = coln + (lane & 15);
      float bias = ba_in[n];
      int half = lane >> 4;
#pragma unroll
      for (int r = 0; r < 8; ++r){
        int m = r + 8*half;
        inp_bf[(b0+m)*512 + n] = f2bf(acc[r] + bias);
      }
    }
    target += 16; group_barrier(barp, target);               // B2

    // ---------- Phase C: att[m, l] = sum_h V[h] tanh(inp + ctx) (l-slice) ----------
    {
      ushort_t* sInp = sA + 8192;
#pragma unroll
      for (int it = 0; it < 4; ++it){
        int id = tid + it*256;
        int row = id >> 6, q = id & 63;
        async_g2l_b128(&sInp[row*512 + q*8], inp_bf + (b0+row)*512 + q*8);
      }
      wait_async();
      __syncthreads();
      int m = tid >> 4;
      int l = w*16 + (tid & 15);
      int b = b0 + m;
      const ushort_t* cp = ctx_bf + ((size_t)b*256 + l)*512;
      float acc = 0.f;
      for (int h = 0; h < 512; h += 8){
        uint4 cq = *(const uint4*)(cp + h);
        uint4 iq = *(const uint4*)(sInp + m*512 + h);
        const ushort_t* cu = (const ushort_t*)&cq;
        const ushort_t* iu = (const ushort_t*)&iq;
#pragma unroll
        for (int j = 0; j < 8; ++j)
          acc += sV[h+j] * ftanh(bf2f(iu[j]) + bf2f(cu[j]));
      }
      float mv = mask_g[b*256 + l];
      att_g[b*256 + l] = (mv == 0.f) ? NEG_BIG : acc;
    }
    target += 16; group_barrier(barp, target);               // B3

    // ---------- Phase D: full softmax (redundant per WG) ----------
    if (tid < 16){
      int b = b0 + tid;
      const float* ar = att_g + b*256;
      float mx = NEG_BIG;
      for (int l = 0; l < 256; ++l) mx = fmaxf(mx, ar[l]);
      float s = 0.f;
      for (int l = 0; l < 256; ++l) s += __expf(ar[l] - mx);
      sRmax[tid] = mx; sRsum[tid] = 1.0f / s;
    }
    __syncthreads();
#pragma unroll
    for (int it = 0; it < 16; ++it){
      int m = tid >> 4;
      int l = it*16 + (tid & 15);
      sRed[m*256 + l] = __expf(att_g[(b0+m)*256 + l] - sRmax[m]) * sRsum[m];
    }
    __syncthreads();
    {
      int m = tid >> 4;
      int l = w*16 + (tid & 15);
      out[(size_t)(b0+m)*65536 + t*256 + l] = sRed[m*256 + l];
    }

    // ---------- Phase E: hs[:, h-slice] = sum_l ctx * alpha ----------
    {
      int h  = hw + (tid & 31);
      int m0 = tid >> 5;                   // uniform per wave
#pragma unroll
      for (int mm = 0; mm < 2; ++mm){
        int m = m0 + mm*8;
        int b = b0 + m;
        const ushort_t* cp = ctx_bf + (size_t)b*256*512 + h;
        float acc = 0.f;
        for (int l = 0; l < 256; ++l)
          acc += sRed[m*256 + l] * bf2f(cp[(size_t)l*512]);
        hs_bf[b*512 + h] = f2bf(acc);
      }
    }
    target += 16; group_barrier(barp, target);               // B4

    // ---------- Phase F: h_t = tanh([hs|h_lstm] @ Who^T + bho) ----------
#pragma unroll
    for (int it = 0; it < 8; ++it){
      int id = tid + it*256;
      int row = id >> 7, q = id & 127;
      int b = b0 + row;
      const ushort_t* src = (q < 64) ? (hs_bf    + b*512 + q*8)
                                     : (hlstm_bf + b*512 + (q-64)*8);
      async_g2l_b128(&sA[row*1024 + q*8], src);
    }
    wait_async();
    __syncthreads();
    if (wv < 2){
      int coln = hw + wv*16;
      const ushort_t* wrow_o = who_bf + (size_t)(coln + (lane & 15)) * 1024 + ((lane >> 4) << 3);
      v8f acc = {0.f,0.f,0.f,0.f,0.f,0.f,0.f,0.f};
      for (int k = 0; k < 32; ++k){
        int kk = k*32;
        if (kk + 128 < 1024) __builtin_prefetch(wrow_o + kk + 128, 0, 3);
        v16bf a = load_frag(sA, 1024, 0, kk);
        v16bf b = load_frag(who_bf, 1024, coln, kk);
        acc = wmma_bf16(a, b, acc);
      }
      int n = coln + (lane & 15);
      float bias = bho[n];
      int half = lane >> 4;
#pragma unroll
      for (int r = 0; r < 8; ++r){
        int m = r + 8*half;
        float ht = ftanh(acc[r] + bias);
        h_bf[(b0+m)*512 + n] = f2bf(ht);
        if (t == LSEQ-1) out[OUT_HF + (b0+m)*512 + n] = ht;
      }
    }

    // ---------- Phase G: pointer select / mask / next dec input (WG0) ----------
    if (w == 0){
      __syncthreads();
      if (tid < 16){
        int m = tid, b = b0 + m;
        float best = -1.0f; int bidx = 0;
        float* mr = mask_g + b*256;
        for (int l = 0; l < 256; ++l){
          float v = sRed[m*256 + l] * mr[l];
          if (v > best){ best = v; bidx = l; }
        }
        sIdx[m] = bidx;
        out[OUT_PTR + b*256 + t] = (float)bidx;
        mr[bidx] = 0.0f;
      }
      __syncthreads();
      {
        int m = tid >> 4;
        int b = b0 + m;
        int j0 = (tid & 15) * 32;
        const float* ep = embedded + ((size_t)b*256 + sIdx[m])*512;
        for (int j = 0; j < 32; ++j)
          dec_bf[b*512 + j0 + j] = f2bf(ep[j0 + j]);
      }
    }
    target += 16; group_barrier(barp, target);               // B5
  }
}

extern "C" void kernel_launch(void* const* d_in, const int* in_sizes, int n_in,
                              void* d_out, int out_size, void* d_ws, size_t ws_size,
                              hipStream_t stream)
{
  (void)in_sizes; (void)n_in; (void)out_size;
  const float* embedded = (const float*)d_in[0];
  const float* dec0     = (const float*)d_in[1];
  const float* h0       = (const float*)d_in[2];
  const float* c0       = (const float*)d_in[3];
  const float* context  = (const float*)d_in[4];
  const float* Wih      = (const float*)d_in[5];
  const float* bih      = (const float*)d_in[6];
  const float* Whh      = (const float*)d_in[7];
  const float* bhh      = (const float*)d_in[8];
  const float* Who      = (const float*)d_in[9];
  const float* bho      = (const float*)d_in[10];
  const float* Wa_in    = (const float*)d_in[11];
  const float* ba_in    = (const float*)d_in[12];
  const float* Wa_ctx   = (const float*)d_in[13];
  const float* ba_ctx   = (const float*)d_in[14];
  const float* V        = (const float*)d_in[15];
  float* out = (float*)d_out;
  char* ws = (char*)d_ws;

  ushort_t* ctx_bf   = (ushort_t*)(ws + CTXBF_OFF);
  ushort_t* wih_bf   = (ushort_t*)(ws + WIH_OFF);
  ushort_t* whh_bf   = (ushort_t*)(ws + WHH_OFF);
  ushort_t* wain_bf  = (ushort_t*)(ws + WAIN_OFF);
  ushort_t* who_bf   = (ushort_t*)(ws + WHO_OFF);
  ushort_t* wactx_bf = (ushort_t*)(ws + WACTX_OFF);
  ushort_t* h_bf     = (ushort_t*)(ws + HBF_OFF);
  ushort_t* hlstm_bf = (ushort_t*)(ws + HLSTM_OFF);
  ushort_t* hs_bf    = (ushort_t*)(ws + HS_OFF);
  ushort_t* dec_bf   = (ushort_t*)(ws + DEC_OFF);
  ushort_t* inp_bf   = (ushort_t*)(ws + INPBF_OFF);
  float*    c_f      = (float*)   (ws + C_OFF);
  float*    mask_g   = (float*)   (ws + MASK_OFF);
  float*    att_g    = (float*)   (ws + ATT_OFF);
  unsigned* bar      = (unsigned*)(ws + BAR_OFF);

  bool big = (ws_size >= WS_NEED_BIG);
  ushort_t* ctxin_bf = big ? (ushort_t*)(ws + CTXIN_OFF) : (ushort_t*)nullptr;

  k_init<<<1024, 256, 0, stream>>>(Wih, Whh, Wa_in, Who, Wa_ctx, h0, c0, dec0, context,
                                   wih_bf, whh_bf, wain_bf, who_bf, wactx_bf,
                                   h_bf, dec_bf, c_f, mask_g, bar, ctxin_bf);
  if (big)
    k_ctx_bf<<<8192, 256, 0, stream>>>(ctxin_bf, wactx_bf, ba_ctx, ctx_bf);
  else
    k_ctx_f32<<<8192, 256, 0, stream>>>(context, wactx_bf, ba_ctx, ctx_bf);
  k_decode<<<128, 256, 0, stream>>>(embedded, bih, bhh, bho, ba_in, V,
                                    ctx_bf, wih_bf, whh_bf, wain_bf, who_bf,
                                    h_bf, hlstm_bf, hs_bf, dec_bf, inp_bf,
                                    c_f, mask_g, att_g, bar, out);
}
